// CRF_layer_41231686041796
// MI455X (gfx1250) — matrix operand order repaired and checked by reading, not compiled, not run on hardware
//
#include <hip/hip_runtime.h>

// CRF log-likelihood for S=512, B=256, T=128 on gfx1250 (MI455X).
//
// Denominator recurrence rewritten as a rescaled GEMM:
//   new[b,j] = emis[s,b,j] + m_b + log( sum_i exp(lh[b,i]-m_b) * exp(trans[i,j]) )
// E = exp(trans) is loop-invariant: each wave keeps its 4 B fragments (32
// VGPRs) in registers for all 512 steps. Per step each wave issues 4 chained
// v_wmma_f32_16x16x32_f16 (16x128 x 128x16 tile). P is packed f16 and written
// with one 16B LDS store per thread; row maxima reduce via __shfl_xor inside
// a 16-lane group (2 barriers/step). Emissions are register double-buffered.
// Transcendentals on the serial critical path use raw v_exp_f32/v_log_f32
// (arguments are provably in normal range, so no denorm fixups needed).

typedef __attribute__((ext_vector_type(16))) _Float16 v16h;
typedef __attribute__((ext_vector_type(8)))  _Float16 v8h;
typedef __attribute__((ext_vector_type(8)))  float    v8f;
typedef __attribute__((ext_vector_type(4)))  float    v4f;

#define S_LEN 512
#define B_SZ  256
#define T_TAG 128
#define MROWS 16     // batch rows per workgroup (WMMA M)
#define NTILES 8     // 128 / 16 N tiles (one per wave)
#define KSTEPS 4     // 128 / 32 K chunks per WMMA chain

// Raw hardware transcendentals (base-2). The trailing v_nop satisfies the
// CDNA5 TRANS-op hazard rule (1 instruction before the result may be used),
// which the compiler cannot enforce across inline asm.
__device__ __forceinline__ float raw_exp2(float x) {
    float r;
    asm("v_exp_f32 %0, %1\n\tv_nop" : "=v"(r) : "v"(x));
    return r;
}
__device__ __forceinline__ float raw_log2(float x) {
    float r;
    asm("v_log_f32 %0, %1\n\tv_nop" : "=v"(r) : "v"(x));
    return r;
}
#define LOG2E_F 1.44269504088896340736f
#define LN2_F   0.69314718055994530942f
__device__ __forceinline__ float fast_exp(float x) { return raw_exp2(x * LOG2E_F); }
__device__ __forceinline__ float fast_log(float x) { return raw_log2(x) * LN2_F; }

// ---------------------------------------------------------------------------
// Forward (log-partition) kernel: grid = B/16 blocks of 256 threads (8 waves)
// ---------------------------------------------------------------------------
__global__ __launch_bounds__(256) void crf_forward_kernel(
    const float* __restrict__ emis, const int* __restrict__ masks,
    const float* __restrict__ st, const float* __restrict__ ed,
    const float* __restrict__ trans, float* __restrict__ denom_out)
{
    // E in WMMA B-fragment layout (staging only; copied to registers): 32 KB
    __shared__ __align__(32) _Float16 E_B[NTILES][KSTEPS][32][16];
    // P in WMMA A-fragment layout: [kchunk][lane][elem]            4 KB
    __shared__ __align__(32) _Float16 P_A[KSTEPS][32][16];
    __shared__ __align__(16) float loglh[MROWS][T_TAG];   // 8 KB state
    __shared__ float rowred[MROWS];                        // per-row max
    __shared__ float mrow[MROWS];                          // per-row mask

    const int tid  = threadIdx.x;
    const int w    = tid >> 5;          // wave index == N tile
    const int lane = tid & 31;
    const int khi  = lane >> 4;         // lane half
    const int nlo  = lane & 15;         // N within tile
    const int jcol = w * 16 + nlo;      // this lane's output column (tag j)
    const int b0   = blockIdx.x * MROWS;

    // reduce mapping: thread owns row rr, 8 consecutive columns [8*cc, 8*cc+8)
    const int rr = tid >> 4;            // row 0..15
    const int cc = tid & 15;            // col group 0..15
    // A-fragment destination for those 8 columns (contiguous 16B chunk):
    const int pa_kk   = cc >> 2;
    const int pa_lane = 16 * ((cc >> 1) & 1) + rr;
    const int pa_e0   = 8 * (cc & 1);

    // ---- build E_B = exp(transitions) directly in B-fragment layout ----
    for (int idx = tid; idx < NTILES * KSTEPS * 32 * 16; idx += 256) {
        int e  = idx & 15;
        int L  = (idx >> 4) & 31;
        int kk = (idx >> 9) & 3;
        int ww = idx >> 11;
        int K  = kk * 32 + ((L >> 4) << 4) + e;   // B layout: K = 16*lanehalf + elem
        int N  = ww * 16 + (L & 15);
        E_B[ww][kk][L][e] = (_Float16)__expf(trans[K * T_TAG + N]);
    }

    // ---- init log_lh = st_transitions + emission[0] ----
    #pragma unroll
    for (int jj = 0; jj < 8; ++jj) {
        int j = 8 * cc + jj;
        loglh[rr][j] = st[j] + emis[(size_t)(b0 + rr) * T_TAG + j];
    }
    __syncthreads();

    // ---- hoist B fragments into registers for the whole recurrence ----
    v16h b_frag[KSTEPS];
    #pragma unroll
    for (int kk = 0; kk < KSTEPS; ++kk)
        b_frag[kk] = *(const v16h*)(&E_B[w][kk][lane][0]);

    // ---- prefetch emission tile for s = 1 (per-lane D-fragment rows) ----
    float ereg[8];
    #pragma unroll
    for (int v = 0; v < 8; ++v) {
        int r = v + 8 * khi;            // D-fragment row for acc[v]
        ereg[v] = emis[(size_t)1 * B_SZ * T_TAG + (size_t)(b0 + r) * T_TAG + jcol];
    }

    for (int s = 1; s < S_LEN; ++s) {
        __syncthreads();                // previous step's loglh writes visible

        // ---- read 8 consecutive log-lh values; row max via 16-lane shuffle ----
        v4f lh0 = *(const v4f*)(&loglh[rr][8 * cc]);
        v4f lh1 = *(const v4f*)(&loglh[rr][8 * cc + 4]);
        float pm = fmaxf(fmaxf(fmaxf(lh0[0], lh0[1]), fmaxf(lh0[2], lh0[3])),
                         fmaxf(fmaxf(lh1[0], lh1[1]), fmaxf(lh1[2], lh1[3])));
        pm = fmaxf(pm, __shfl_xor(pm, 1));
        pm = fmaxf(pm, __shfl_xor(pm, 2));
        pm = fmaxf(pm, __shfl_xor(pm, 4));
        pm = fmaxf(pm, __shfl_xor(pm, 8));
        if (cc == 0) rowred[rr] = pm;                       // for update stage
        if (tid < MROWS) mrow[tid] = (float)masks[s * B_SZ + b0 + tid];

        // ---- P = exp(loglh - rowmax) packed f16, one 16B store per thread ----
        v8h pv;
        pv[0] = (_Float16)fast_exp(lh0[0] - pm);
        pv[1] = (_Float16)fast_exp(lh0[1] - pm);
        pv[2] = (_Float16)fast_exp(lh0[2] - pm);
        pv[3] = (_Float16)fast_exp(lh0[3] - pm);
        pv[4] = (_Float16)fast_exp(lh1[0] - pm);
        pv[5] = (_Float16)fast_exp(lh1[1] - pm);
        pv[6] = (_Float16)fast_exp(lh1[2] - pm);
        pv[7] = (_Float16)fast_exp(lh1[3] - pm);
        *(v8h*)(&P_A[pa_kk][pa_lane][pa_e0]) = pv;
        __syncthreads();

        // ---- GEMM: acc(16x16) = P(16x128) x E(128x16 tile w) ----
        v8f acc = {};
        #pragma unroll
        for (int kk = 0; kk < KSTEPS; ++kk) {
            v16h a = *(const v16h*)(&P_A[kk][lane][0]);
            acc = __builtin_amdgcn_wmma_f32_16x16x32_f16(
                      false, a, false, b_frag[kk], (short)0, acc, false, false);
        }

        // ---- update: loglh = mask * (emis + max + log(acc)) + (1-mask)*old ----
        #pragma unroll
        for (int v = 0; v < 8; ++v) {
            int r = v + 8 * khi;
            float newv = ereg[v] + rowred[r] + fast_log(acc[v]);
            float m    = mrow[r];
            float oldv = loglh[r][jcol];
            loglh[r][jcol] = m * newv + (1.0f - m) * oldv;
        }

        // ---- prefetch next step's emissions (overlaps next iteration) ----
        if (s + 1 < S_LEN) {
            #pragma unroll
            for (int v = 0; v < 8; ++v) {
                int r = v + 8 * khi;
                ereg[v] = emis[(size_t)(s + 1) * B_SZ * T_TAG +
                               (size_t)(b0 + r) * T_TAG + jcol];
            }
        }
    }

    // ---- denom[b] = logsumexp_j(loglh[b,j] + ed[j]) ----
    __syncthreads();
    {
        v4f lh0 = *(const v4f*)(&loglh[rr][8 * cc]);
        v4f lh1 = *(const v4f*)(&loglh[rr][8 * cc + 4]);
        float val[8];
        #pragma unroll
        for (int i = 0; i < 4; ++i) {
            val[i]     = lh0[i] + ed[8 * cc + i];
            val[i + 4] = lh1[i] + ed[8 * cc + 4 + i];
        }
        float pm = val[0];
        #pragma unroll
        for (int i = 1; i < 8; ++i) pm = fmaxf(pm, val[i]);
        pm = fmaxf(pm, __shfl_xor(pm, 1));
        pm = fmaxf(pm, __shfl_xor(pm, 2));
        pm = fmaxf(pm, __shfl_xor(pm, 4));
        pm = fmaxf(pm, __shfl_xor(pm, 8));
        float ps = 0.0f;
        #pragma unroll
        for (int i = 0; i < 8; ++i) ps += fast_exp(val[i] - pm);
        ps += __shfl_xor(ps, 1);
        ps += __shfl_xor(ps, 2);
        ps += __shfl_xor(ps, 4);
        ps += __shfl_xor(ps, 8);
        if (cc == 0) denom_out[b0 + rr] = pm + fast_log(ps);
    }
}

// ---------------------------------------------------------------------------
// Numerator (gold-path score): one block per batch element, s split over lanes
// ---------------------------------------------------------------------------
__global__ __launch_bounds__(128) void crf_num_kernel(
    const float* __restrict__ emis, const int* __restrict__ tags,
    const int* __restrict__ masks, const float* __restrict__ st,
    const float* __restrict__ ed, const float* __restrict__ trans,
    float* __restrict__ num_out)
{
    __shared__ float sred[128];
    __shared__ int   ired[128];
    const int b = blockIdx.x;
    const int t = threadIdx.x;

    float partial = 0.0f;
    int mcount = 0;
    for (int s = t; s < S_LEN; s += 128) {
        int tag = tags[s * B_SZ + b];
        int m   = masks[s * B_SZ + b];
        float mf = (float)m;
        if (s < S_LEN - 1)
            partial += emis[(size_t)s * B_SZ * T_TAG + (size_t)b * T_TAG + tag] * mf;
        if (s >= 1) {
            int tprev = tags[(s - 1) * B_SZ + b];
            partial += trans[tprev * T_TAG + tag] * mf;
        }
        mcount += m;
    }
    sred[t] = partial;
    ired[t] = mcount;
    __syncthreads();
    for (int off = 64; off > 0; off >>= 1) {
        if (t < off) { sred[t] += sred[t + off]; ired[t] += ired[t + off]; }
        __syncthreads();
    }
    if (t == 0) {
        int last_idx = ired[0] - 1;
        int tag0     = tags[b];
        int last_tag = tags[last_idx * B_SZ + b];
        float mlast  = (float)masks[(S_LEN - 1) * B_SZ + b];
        num_out[b] = st[tag0] + sred[0] + ed[last_tag] +
                     emis[(size_t)(S_LEN - 1) * B_SZ * T_TAG +
                          (size_t)b * T_TAG + last_tag] * mlast;
    }
}

// ---------------------------------------------------------------------------
// Final reduction: sum_b (num[b] - denom[b]) -> scalar
// ---------------------------------------------------------------------------
__global__ __launch_bounds__(256) void crf_reduce_kernel(
    const float* __restrict__ num, const float* __restrict__ denom,
    float* __restrict__ out)
{
    __shared__ float sred[256];
    int t = threadIdx.x;
    sred[t] = num[t] - denom[t];
    __syncthreads();
    for (int off = 128; off > 0; off >>= 1) {
        if (t < off) sred[t] += sred[t + off];
        __syncthreads();
    }
    if (t == 0) out[0] = sred[0];
}

// ---------------------------------------------------------------------------
extern "C" void kernel_launch(void* const* d_in, const int* in_sizes, int n_in,
                              void* d_out, int out_size, void* d_ws, size_t ws_size,
                              hipStream_t stream)
{
    const float* emis  = (const float*)d_in[0];   // [S,B,T] f32
    const int*   tags  = (const int*)  d_in[1];   // [S,B] i32
    const int*   masks = (const int*)  d_in[2];   // [S,B] i32
    const float* st    = (const float*)d_in[3];   // [T] f32
    const float* ed    = (const float*)d_in[4];   // [T] f32
    const float* trans = (const float*)d_in[5];   // [T,T] f32
    float* out = (float*)d_out;

    float* ws_num   = (float*)d_ws;        // [B]
    float* ws_denom = ws_num + B_SZ;       // [B]

    crf_num_kernel<<<B_SZ, 128, 0, stream>>>(emis, tags, masks, st, ed, trans, ws_num);
    crf_forward_kernel<<<B_SZ / MROWS, 256, 0, stream>>>(emis, masks, st, ed, trans, ws_denom);
    crf_reduce_kernel<<<1, 256, 0, stream>>>(ws_num, ws_denom, out);
}
